// REWAMultiHeadAttention_46557445489425
// MI455X (gfx1250) — matrix operand order repaired and checked by reading, not compiled, not run on hardware
//
#include <hip/hip_runtime.h>
#include <math.h>

typedef __attribute__((ext_vector_type(16))) _Float16 v16h;
typedef __attribute__((ext_vector_type(8)))  _Float16 v8h;
typedef __attribute__((ext_vector_type(8)))  float    v8f;

#define NB 2
#define NN 1024
#define ND 1024
#define NH 16
#define HD 64
#define FEPS 1e-6f

// ---------- helpers ----------
__device__ __forceinline__ v16h cat16(v8h lo, v8h hi) {
  return __builtin_shufflevector(lo, hi, 0,1,2,3,4,5,6,7,8,9,10,11,12,13,14,15);
}
// Load A-operand (16x32 f16, M x K) slice for this lane from a row-major row
// pointer p (start of the 32-wide K chunk). Lane layout per ISA: lane holds
// K = {s*8..s*8+7, 16+s*8..16+s*8+7}, s = lane>>4.
__device__ __forceinline__ v16h loadA32(const _Float16* p, int s) {
  v8h lo = *(const v8h*)(p + s * 8);
  v8h hi = *(const v8h*)(p + 16 + s * 8);
  return cat16(lo, hi);
}
__device__ __forceinline__ v8f wmma16(v16h a, v16h b, v8f c) {
  return __builtin_amdgcn_wmma_f32_16x16x32_f16(false, a, false, b, (short)0, c,
                                                false, false);
}
__device__ __forceinline__ float wredsum(float v) {
#pragma unroll
  for (int i = 16; i > 0; i >>= 1) v += __shfl_xor(v, i, 32);
  return v;
}
// REWA combined-metric logit for one score element.
__device__ __forceinline__ float score_elem(float innE, float innS, float innH,
                                            float innF, float4 ks, float sqEq,
                                            float sqHq, float omHq, float sqFq,
                                            float al, float be, float ga, float de,
                                            float invT, int mk) {
  float dE = sqrtf(fmaxf(sqEq + ks.x - 2.0f * innE, 0.0f) + FEPS);
  float cv = fminf(fmaxf(innS, -1.0f + FEPS), 1.0f - FEPS);
  float dS = acosf(cv);
  float d2H = fmaxf(sqHq + ks.y - 2.0f * innH, 0.0f);
  float arg = fmaxf(1.0f + 2.0f * d2H / (omHq * ks.z + FEPS), 1.0f + FEPS);
  float dH = __logf(arg + sqrtf(arg * arg - 1.0f));
  float dF = sqrtf(fmaxf(sqFq + ks.w - 2.0f * innF, 0.0f) + FEPS);
  float dist = (al * dE + be * dS + ga * dH + de * dF) * invT;
  return (mk > 0) ? -dist : -1.0e9f;
}

// ---------- K0: fp32 -> f16 conversions ----------
__global__ void cvt_kernel(const float* __restrict__ x, const float* __restrict__ Wv,
                           const float* __restrict__ Wo, _Float16* __restrict__ x16,
                           _Float16* __restrict__ Wv16, _Float16* __restrict__ Wo16) {
  int i = blockIdx.x * 256 + threadIdx.x;
  if (i < NB * NN * ND) x16[i] = (_Float16)x[i];
  if (i < ND * ND) {
    Wv16[i] = (_Float16)Wv[i];
    Wo16[i] = (_Float16)Wo[i];
  }
}

// ---------- K1: per-row witness features (one wave per (b,h,n) row) ----------
__global__ void feat_kernel(const float* __restrict__ x, const float* __restrict__ WE,
                            const float* __restrict__ bE, const float* __restrict__ WS,
                            const float* __restrict__ bS, const float* __restrict__ WH,
                            const float* __restrict__ bH, const float* __restrict__ Wr,
                            const float* __restrict__ br, _Float16* __restrict__ feat,
                            float* __restrict__ sq4) {
  __shared__ float xs[8][64];
  int lane = threadIdx.x & 31, wid = threadIdx.x >> 5;
  int row = blockIdx.x * 8 + wid;            // [0, B*H*N)
  int bh = row / NN, n = row % NN;
  int b = bh / NH, h = bh % NH;

  const float* xr = x + ((size_t)(b * NN + n)) * ND + h * HD;
  xs[wid][lane] = xr[lane];
  xs[wid][lane + 32] = xr[lane + 32];
  __syncthreads();

  const float* we = WE + h * HD * 64;
  const float* wss = WS + h * HD * 64;
  const float* wh = WH + h * HD * 64;
  const float* wr = Wr + h * HD * 128;
  float aE0 = 0, aE1 = 0, aS0 = 0, aS1 = 0, aH0 = 0, aH1 = 0;
  float aF0 = 0, aF1 = 0, aF2 = 0, aF3 = 0;
#pragma unroll 4
  for (int k = 0; k < 64; ++k) {
    float xv = xs[wid][k];
    aE0 += xv * we[k * 64 + lane];       aE1 += xv * we[k * 64 + lane + 32];
    aS0 += xv * wss[k * 64 + lane];      aS1 += xv * wss[k * 64 + lane + 32];
    aH0 += xv * wh[k * 64 + lane];       aH1 += xv * wh[k * 64 + lane + 32];
    aF0 += xv * wr[k * 128 + lane];      aF1 += xv * wr[k * 128 + lane + 32];
    aF2 += xv * wr[k * 128 + lane + 64]; aF3 += xv * wr[k * 128 + lane + 96];
  }
  aE0 += bE[h * 64 + lane]; aE1 += bE[h * 64 + lane + 32];
  aS0 += bS[h * 64 + lane]; aS1 += bS[h * 64 + lane + 32];
  aH0 += bH[h * 64 + lane]; aH1 += bH[h * 64 + lane + 32];

  float sqE = wredsum(aE0 * aE0 + aE1 * aE1);
  float s2 = wredsum(aS0 * aS0 + aS1 * aS1);
  float invS = 1.0f / (sqrtf(s2) + FEPS);
  float u2 = wredsum(aH0 * aH0 + aH1 * aH1);
  float un = sqrtf(u2);
  float sc = tanhf(un) / (un + FEPS);
  float sqHr = sc * sc * u2;
  float omsq = 1.0f - fminf(fmaxf(sqHr, 0.0f), 1.0f - 1e-5f);
  const float zk = 0.125f;  // sqrt(2/128)
  float z0 = zk * __cosf(aF0 + br[h * 128 + lane]);
  float z1 = zk * __cosf(aF1 + br[h * 128 + lane + 32]);
  float z2 = zk * __cosf(aF2 + br[h * 128 + lane + 64]);
  float z3 = zk * __cosf(aF3 + br[h * 128 + lane + 96]);
  float sqF = wredsum(z0 * z0 + z1 * z1 + z2 * z2 + z3 * z3);

  _Float16* fr = feat + (size_t)row * 320;
  fr[lane] = (_Float16)aE0;            fr[lane + 32] = (_Float16)aE1;
  fr[64 + lane] = (_Float16)(aS0 * invS);  fr[64 + lane + 32] = (_Float16)(aS1 * invS);
  fr[128 + lane] = (_Float16)(aH0 * sc);   fr[128 + lane + 32] = (_Float16)(aH1 * sc);
  fr[192 + lane] = (_Float16)z0;       fr[192 + lane + 32] = (_Float16)z1;
  fr[192 + lane + 64] = (_Float16)z2;  fr[192 + lane + 96] = (_Float16)z3;
  if (lane == 0) ((float4*)sq4)[row] = make_float4(sqE, sqHr, omsq, sqF);
}

// ---------- K2: v = x @ Wv^T + bv, stored transposed vT[b][h][d][n] f16 ----------
__global__ void vproj_kernel(const _Float16* __restrict__ x16,
                             const _Float16* __restrict__ Wv16,
                             const float* __restrict__ bv, _Float16* __restrict__ vT) {
  int lane = threadIdx.x & 31, wid = threadIdx.x >> 5;
  int tile = blockIdx.x * 4 + wid;  // 8192 tiles = 128 row-tiles x 64 col-tiles
  int rt = tile >> 6, jt = tile & 63;
  int l16 = lane & 15, s = lane >> 4;
  const _Float16* arow = x16 + (size_t)(rt * 16 + l16) * ND;
  int j = jt * 16 + l16;
  const _Float16* brow = Wv16 + (size_t)j * ND;
  v8f c;
#pragma unroll
  for (int r = 0; r < 8; ++r) c[r] = 0.0f;
  for (int k0 = 0; k0 < ND; k0 += 32) {
    v16h a = loadA32(arow + k0, s);
    v16h bm = loadA32(brow + k0, s);
    c = wmma16(a, bm, c);
  }
  float bias = bv[j];
  v8h o;
#pragma unroll
  for (int r = 0; r < 8; ++r) o[r] = (_Float16)(c[r] + bias);
  int token0 = rt * 16 + 8 * s;
  int bb = token0 / NN, n0 = token0 % NN;
  int h = j >> 6, d = j & 63;
  *(v8h*)(vT + (((size_t)(bb * NH + h) * 64 + d) * NN + n0)) = o;
}

// ---------- K3: fused multi-metric flash attention (32 keys / iteration) ----------
__global__ void attn_kernel(const _Float16* __restrict__ feat,
                            const float* __restrict__ sq4,
                            const _Float16* __restrict__ vT, const int* __restrict__ mask,
                            const float* __restrict__ alpha, const float* __restrict__ beta,
                            const float* __restrict__ gamma, const float* __restrict__ delta,
                            const float* __restrict__ temp, _Float16* __restrict__ O16) {
  __shared__ __attribute__((aligned(64))) _Float16 PT[4][16][32];
  int lane = threadIdx.x & 31, wid = threadIdx.x >> 5;
  int tile = blockIdx.x * 4 + wid;  // 2048 tiles = (B*H) x (N/16)
  int bh = tile >> 6, qt = tile & 63;
  int b = bh >> 4, h = bh & 15;
  int l16 = lane & 15, s = lane >> 4;

  // query-side B operands: lane holds column q=l16, e = {s*8..s*8+7, 16+s*8..16+s*8+7}
  const _Float16* qb = feat + ((size_t)bh * NN + qt * 16 + l16) * 320;
  v16h qE0 = loadA32(qb + 0, s),   qE1 = loadA32(qb + 32, s);
  v16h qS0 = loadA32(qb + 64, s),  qS1 = loadA32(qb + 96, s);
  v16h qH0 = loadA32(qb + 128, s), qH1 = loadA32(qb + 160, s);
  v16h qF0 = loadA32(qb + 192, s), qF1 = loadA32(qb + 224, s);
  v16h qF2 = loadA32(qb + 256, s), qF3 = loadA32(qb + 288, s);

  float4 qs = ((const float4*)sq4)[(size_t)bh * NN + qt * 16 + l16];
  float sqEq = qs.x, sqHq = qs.y, omHq = qs.z, sqFq = qs.w;
  float al = alpha[h], be = beta[h], ga = gamma[h], de = delta[h];
  float invT = 1.0f / temp[h];

  float m = -3.0e38f, lsum = 0.0f;
  v8f o[4];
#pragma unroll
  for (int dt = 0; dt < 4; ++dt)
#pragma unroll
    for (int r = 0; r < 8; ++r) o[dt][r] = 0.0f;

  const _Float16* vbase = vT + (size_t)bh * 64 * NN;
  const float4* ksq4 = ((const float4*)sq4) + (size_t)bh * NN;
  const int* mrow = mask + b * NN;

  for (int kt2 = 0; kt2 < 32; ++kt2) {
    int kb = kt2 * 32;  // 32 keys: tile A = kb..kb+15, tile B = kb+16..kb+31
    float lgA[8], lgB[8];

    // ---- score tile A ----
    {
      const _Float16* kf = feat + ((size_t)bh * NN + kb + l16) * 320;
      v8f cE, cS, cH, cF;
#pragma unroll
      for (int r = 0; r < 8; ++r) { cE[r] = 0; cS[r] = 0; cH[r] = 0; cF[r] = 0; }
      cE = wmma16(loadA32(kf + 0, s), qE0, cE);
      cE = wmma16(loadA32(kf + 32, s), qE1, cE);
      cS = wmma16(loadA32(kf + 64, s), qS0, cS);
      cS = wmma16(loadA32(kf + 96, s), qS1, cS);
      cH = wmma16(loadA32(kf + 128, s), qH0, cH);
      cH = wmma16(loadA32(kf + 160, s), qH1, cH);
      cF = wmma16(loadA32(kf + 192, s), qF0, cF);
      cF = wmma16(loadA32(kf + 224, s), qF1, cF);
      cF = wmma16(loadA32(kf + 256, s), qF2, cF);
      cF = wmma16(loadA32(kf + 288, s), qF3, cF);
#pragma unroll
      for (int r = 0; r < 8; ++r) {
        int k = kb + 8 * s + r;
        lgA[r] = score_elem(cE[r], cS[r], cH[r], cF[r], ksq4[k], sqEq, sqHq, omHq,
                            sqFq, al, be, ga, de, invT, mrow[k]);
      }
    }
    // ---- score tile B ----
    {
      const _Float16* kf = feat + ((size_t)bh * NN + kb + 16 + l16) * 320;
      v8f cE, cS, cH, cF;
#pragma unroll
      for (int r = 0; r < 8; ++r) { cE[r] = 0; cS[r] = 0; cH[r] = 0; cF[r] = 0; }
      cE = wmma16(loadA32(kf + 0, s), qE0, cE);
      cE = wmma16(loadA32(kf + 32, s), qE1, cE);
      cS = wmma16(loadA32(kf + 64, s), qS0, cS);
      cS = wmma16(loadA32(kf + 96, s), qS1, cS);
      cH = wmma16(loadA32(kf + 128, s), qH0, cH);
      cH = wmma16(loadA32(kf + 160, s), qH1, cH);
      cF = wmma16(loadA32(kf + 192, s), qF0, cF);
      cF = wmma16(loadA32(kf + 224, s), qF1, cF);
      cF = wmma16(loadA32(kf + 256, s), qF2, cF);
      cF = wmma16(loadA32(kf + 288, s), qF3, cF);
#pragma unroll
      for (int r = 0; r < 8; ++r) {
        int k = kb + 16 + 8 * s + r;
        lgB[r] = score_elem(cE[r], cS[r], cH[r], cF[r], ksq4[k], sqEq, sqHq, omHq,
                            sqFq, al, be, ga, de, invT, mrow[k]);
      }
    }

    // ---- combined online softmax over 32 keys ----
    float tm = lgA[0];
#pragma unroll
    for (int r = 1; r < 8; ++r) tm = fmaxf(tm, lgA[r]);
#pragma unroll
    for (int r = 0; r < 8; ++r) tm = fmaxf(tm, lgB[r]);
    tm = fmaxf(tm, __shfl_xor(tm, 16, 32));
    float mnew = fmaxf(m, tm);
    float corr = __expf(m - mnew);
    float psum = 0.0f;
    v8h phA, phB;
#pragma unroll
    for (int r = 0; r < 8; ++r) {
      float pA = __expf(lgA[r] - mnew);
      float pB = __expf(lgB[r] - mnew);
      psum += pA + pB;
      phA[r] = (_Float16)pA;
      phB[r] = (_Float16)pB;
    }
    psum += __shfl_xor(psum, 16, 32);
    lsum = lsum * corr + psum;
#pragma unroll
    for (int dt = 0; dt < 4; ++dt)
#pragma unroll
      for (int r = 0; r < 8; ++r) o[dt][r] *= corr;
    m = mnew;

    // stage P^T (16 q x 32 k) in LDS: lane col q=l16, contiguous key octets
    *(v8h*)&PT[wid][l16][8 * s] = phA;
    *(v8h*)&PT[wid][l16][16 + 8 * s] = phB;
    __builtin_amdgcn_wave_barrier();
    v16h pb = loadA32(&PT[wid][l16][0], s);  // B2 operand, fully dense

#pragma unroll
    for (int dt = 0; dt < 4; ++dt) {
      const _Float16* vr = vbase + (size_t)(dt * 16 + l16) * NN + kb;
      o[dt] = wmma16(loadA32(vr, s), pb, o[dt]);
    }
  }

  float invl = 1.0f / lsum;
  int q = qt * 16 + l16;
  _Float16* ob = O16 + ((size_t)(b * NN + q)) * ND + h * 64;
#pragma unroll
  for (int dt = 0; dt < 4; ++dt) {
    v8h ov;
#pragma unroll
    for (int r = 0; r < 8; ++r) ov[r] = (_Float16)(o[dt][r] * invl);
    *(v8h*)(ob + dt * 16 + 8 * s) = ov;
  }
}

// ---------- K4: out = O @ Wo^T + bo (f32 output) ----------
__global__ void outproj_kernel(const _Float16* __restrict__ O16,
                               const _Float16* __restrict__ Wo16,
                               const float* __restrict__ bo, float* __restrict__ out) {
  int lane = threadIdx.x & 31, wid = threadIdx.x >> 5;
  int tile = blockIdx.x * 4 + wid;
  int rt = tile >> 6, jt = tile & 63;
  int l16 = lane & 15, s = lane >> 4;
  const _Float16* arow = O16 + (size_t)(rt * 16 + l16) * ND;
  int j = jt * 16 + l16;
  const _Float16* brow = Wo16 + (size_t)j * ND;
  v8f c;
#pragma unroll
  for (int r = 0; r < 8; ++r) c[r] = 0.0f;
  for (int k0 = 0; k0 < ND; k0 += 32) {
    v16h a = loadA32(arow + k0, s);
    v16h bm = loadA32(brow + k0, s);
    c = wmma16(a, bm, c);
  }
  float bias = bo[j];
  int token0 = rt * 16 + 8 * s;
#pragma unroll
  for (int r = 0; r < 8; ++r) out[(size_t)(token0 + r) * ND + j] = c[r] + bias;
}

extern "C" void kernel_launch(void* const* d_in, const int* in_sizes, int n_in,
                              void* d_out, int out_size, void* d_ws, size_t ws_size,
                              hipStream_t stream) {
  const float* x = (const float*)d_in[0];
  const int* mask = (const int*)d_in[1];
  const float* WE = (const float*)d_in[2];
  const float* bE = (const float*)d_in[3];
  const float* WS = (const float*)d_in[4];
  const float* bS = (const float*)d_in[5];
  const float* WH = (const float*)d_in[6];
  const float* bH = (const float*)d_in[7];
  const float* Wr = (const float*)d_in[8];
  const float* br = (const float*)d_in[9];
  const float* alpha = (const float*)d_in[10];
  const float* beta = (const float*)d_in[11];
  const float* gamma = (const float*)d_in[12];
  const float* delta = (const float*)d_in[13];
  const float* temp = (const float*)d_in[14];
  const float* Wv = (const float*)d_in[15];
  const float* bv = (const float*)d_in[16];
  const float* Wo = (const float*)d_in[17];
  const float* bo = (const float*)d_in[18];
  float* out = (float*)d_out;

  char* w = (char*)d_ws;  // ~40 MB used
  _Float16* x16 = (_Float16*)(w + 0);          //  4 MB
  _Float16* Wv16 = (_Float16*)(w + 4194304);   //  2 MB
  _Float16* Wo16 = (_Float16*)(w + 6291456);   //  2 MB
  _Float16* feat = (_Float16*)(w + 8388608);   // 20 MB (32768 rows x 320 f16)
  float* sq4 = (float*)(w + 29360128);         //  2 MB (float4 per row)
  _Float16* vT = (_Float16*)(w + 31457280);    //  4 MB
  _Float16* O16 = (_Float16*)(w + 35651840);   //  4 MB

  cvt_kernel<<<8192, 256, 0, stream>>>(x, Wv, Wo, x16, Wv16, Wo16);
  feat_kernel<<<4096, 256, 0, stream>>>(x, WE, bE, WS, bS, WH, bH, Wr, br, feat, sq4);
  vproj_kernel<<<2048, 128, 0, stream>>>(x16, Wv16, bv, vT);
  attn_kernel<<<512, 128, 0, stream>>>(feat, sq4, vT, mask, alpha, beta, gamma, delta,
                                       temp, O16);
  outproj_kernel<<<2048, 128, 0, stream>>>(O16, Wo16, bo, out);
}
